// TnnBinConvBNReLU_14035953123865
// MI455X (gfx1250) — 8-bit matrix operand order repaired and checked by reading, compiled, not run on hardware
//
#include <hip/hip_runtime.h>
#include <hip/hip_bf16.h>
#include <stdint.h>

typedef int v8i __attribute__((ext_vector_type(8)));
typedef unsigned int ui4 __attribute__((ext_vector_type(4)));
typedef int i4v __attribute__((ext_vector_type(4)));

#define B_    32
#define CIN   256
#define COUT  256
#define HS    56
#define WSP   56
#define HP    58
#define WP    58
#define HW    (HS*WSP)
#define M_TOT (B_*HW)
#define KTOT  (CIN*9)
#define BTILE_BYTES (16*KTOT)
#define BTILE_LDS   (BTILE_BYTES + 16*(BTILE_BYTES/1024))
#define BN_EPS 1e-5f

#if __has_builtin(__builtin_amdgcn_tensor_load_to_lds)
#define HAVE_TDM 1
#else
#define HAVE_TDM 0
#endif

__device__ __forceinline__ int posA64(int c) {
  int h = (c >> 4) & 1;
  int v = ((c >> 2) & 3) | (((c >> 5) & 1) << 2);
  return h * 32 + v * 4 + (c & 3);
}
__device__ __forceinline__ int posB64(int c) {
  int h = (c >> 4) & 1;
  int v = ((c >> 2) & 3) | (((c >> 5) & 1) << 2);
  return h * 32 + v * 4 + (c & 3);
}

__global__ void k_prep_w(const float* __restrict__ w,
                         signed char* __restrict__ bq,
                         float* __restrict__ alpha) {
  __shared__ float sred[256];
  const int o = blockIdx.x;
  const int i = threadIdx.x;
  const int kslot = (i >> 6) * 64 + posB64(i & 63);
  float absacc = 0.f;
  for (int p = 0; p < 9; ++p) {
    float wv = w[((size_t)(o * CIN + i)) * 9 + p];
    sred[i] = wv; __syncthreads();
    for (int s = 128; s > 0; s >>= 1) {
      if (i < s) sred[i] += sred[i + s];
      __syncthreads();
    }
    float mean = sred[0] * (1.0f / 256.0f);
    __syncthreads();
    float wc  = wv - mean;
    float wcc = fminf(1.f, fmaxf(-1.f, wc));
    absacc += fabsf(wcc);
    bq[(size_t)o * KTOT + p * CIN + kslot] =
        (signed char)((wc > 0.f) ? 1 : ((wc < 0.f) ? -1 : 0));
  }
  sred[i] = absacc; __syncthreads();
  for (int s = 128; s > 0; s >>= 1) {
    if (i < s) sred[i] += sred[i + s];
    __syncthreads();
  }
  if (i == 0) alpha[o] = sred[0] * (1.0f / (float)KTOT);
}

__global__ void k_bin_x(const float* __restrict__ x, signed char* __restrict__ xq) {
  size_t idx = (size_t)blockIdx.x * blockDim.x + threadIdx.x;
  if (idx >= (size_t)B_ * CIN * HW) return;
  int hw = (int)(idx % HW);
  int c  = (int)((idx / HW) % CIN);
  int b  = (int)(idx / ((size_t)HW * CIN));
  int h = hw / WSP, w = hw % WSP;
  float v = x[idx];
  size_t dst = (((size_t)b * HP + (h + 1)) * WP + (w + 1)) * CIN
             + (c >> 6) * 64 + posA64(c & 63);
  xq[dst] = (signed char)((v > 0.f) ? 1 : ((v < 0.f) ? -1 : 0));
}

__global__ __launch_bounds__(256) void k_conv(
    const signed char* __restrict__ xq,
    const signed char* __restrict__ bq,
    const float* __restrict__ alpha,
    const float* __restrict__ bias,
    float* __restrict__ y) {
  __shared__ signed char btile[BTILE_LDS];

  const int lane = threadIdx.x & 31;
  const int wave = threadIdx.x >> 5;
  const int half = lane >> 4;
  const int n0 = blockIdx.x * 16;
  const int m_base = blockIdx.y * 128 + wave * 16;

#if HAVE_TDM
  if (threadIdx.x < 32) {
    unsigned long long ga = (unsigned long long)(bq + (size_t)n0 * KTOT);
    ui4 g0;
    g0[0] = 1u;
    g0[1] = 0u;
    g0[2] = (unsigned)ga;
    g0[3] = (unsigned)(ga >> 32) | 0x80000000u;
    v8i g1;
    g1[0] = (1 << 20) | (7 << 22) | (3 << 25);
    g1[1] = (int)0x90000000;
    g1[2] = 0x00010000;
    g1[3] = (int)0x90000000;
    g1[4] = 1;
    g1[5] = BTILE_BYTES;
    g1[6] = 0;
    g1[7] = 0;
    i4v z4 = {0, 0, 0, 0};
    v8i z8 = {0, 0, 0, 0, 0, 0, 0, 0};
    __builtin_amdgcn_tensor_load_to_lds(g0, g1, z4, z4, z8, 0);
    __builtin_amdgcn_s_wait_tensorcnt(0);
  }
#else
  {
    const int4* src = (const int4*)(bq + (size_t)n0 * KTOT);
    for (int t = threadIdx.x; t < BTILE_BYTES / 16; t += 256) {
      *(int4*)(btile + (size_t)16 * (t + (t >> 6))) = src[t];
    }
  }
#endif
  __syncthreads();
  asm volatile("" : : "r"(&btile[0]) : "memory");

  const int mrow = m_base + (lane & 15);
  const int b  = mrow / HW;
  const int hw = mrow % HW;
  const int h  = hw / WSP;
  const int w  = hw % WSP;

  const signed char* apix =
      xq + (((size_t)b * HP + (h + 1)) * WP + (w + 1)) * CIN + half * 32;
  const int sbase = (lane & 15) * KTOT + half * 32;

  v8i acc = {0, 0, 0, 0, 0, 0, 0, 0};
#pragma unroll
  for (int p = 0; p < 9; ++p) {
    const int dh = p / 3 - 1, dw = p % 3 - 1;
    const signed char* atap = apix + (ptrdiff_t)(dh * WP + dw) * CIN;
    if (p < 8) {
      const int dh2 = (p + 1) / 3 - 1, dw2 = (p + 1) % 3 - 1;
      __builtin_prefetch(apix + (ptrdiff_t)(dh2 * WP + dw2) * CIN, 0, 3);
    }
#pragma unroll
    for (int ch = 0; ch < 4; ++ch) {
      const int s = sbase + p * 256 + ch * 64;
      const signed char* bptr = btile + s + 16 * (s >> 10);
      const int4 alo = *(const int4*)(atap + ch * 64);
      const int4 ahi = *(const int4*)(atap + ch * 64 + 16);
      const int4 blo = *(const int4*)(bptr);
      const int4 bhi = *(const int4*)(bptr + 16);
      v8i a, bf;
      a[0] = alo.x; a[1] = alo.y; a[2] = alo.z; a[3] = alo.w;
      a[4] = ahi.x; a[5] = ahi.y; a[6] = ahi.z; a[7] = ahi.w;
      bf[0] = blo.x; bf[1] = blo.y; bf[2] = blo.z; bf[3] = blo.w;
      bf[4] = bhi.x; bf[5] = bhi.y; bf[6] = bhi.z; bf[7] = bhi.w;
      acc = __builtin_amdgcn_wmma_i32_16x16x64_iu8(
          true, a, true, bf, acc, false, false);
    }
  }

  const int nb = n0 + (lane & 15);
  const float al = alpha[nb], bs = bias[nb];
  float* yo = y + (size_t)(m_base + half * 8) * COUT + nb;
#pragma unroll
  for (int r = 0; r < 8; ++r)
    yo[(size_t)r * COUT] = al * (float)acc[r] + bs;
}

__global__ void k_stats(const float* __restrict__ y,
                        float* __restrict__ gsum, float* __restrict__ gsq) {
  const int c = threadIdx.x;
  const int m0 = blockIdx.x * 256;
  float s = 0.f, q = 0.f;
  for (int r = 0; r < 256; ++r) {
    float v = y[(size_t)(m0 + r) * COUT + c];
    s += v; q += v * v;
  }
  atomicAdd(&gsum[c], s);
  atomicAdd(&gsq[c], q);
}

__global__ void k_final(const float* __restrict__ y,
                        const float* __restrict__ gsum, const float* __restrict__ gsq,
                        const float* __restrict__ gamma, const float* __restrict__ beta,
                        float* __restrict__ out) {
  size_t idx = (size_t)blockIdx.x * blockDim.x + threadIdx.x;
  if (idx >= (size_t)B_ * COUT * HW) return;
  int hw = (int)(idx % HW);
  int c  = (int)((idx / HW) % COUT);
  int b  = (int)(idx / ((size_t)HW * COUT));
  const float inv = 1.0f / (float)M_TOT;
  float mu  = gsum[c] * inv;
  float var = gsq[c] * inv - mu * mu;
  float v = y[((size_t)b * HW + hw) * COUT + c];
  float t = gamma[c] * (v - mu) * rsqrtf(var + BN_EPS) + beta[c];
  out[idx] = (t > 0.f) ? 1.0f : 0.0f;
}

extern "C" void kernel_launch(void* const* d_in, const int* in_sizes, int n_in,
                              void* d_out, int out_size, void* d_ws, size_t ws_size,
                              hipStream_t stream) {
  const float* x     = (const float*)d_in[0];
  const float* w     = (const float*)d_in[1];
  const float* bias  = (const float*)d_in[2];
  const float* gamma = (const float*)d_in[3];
  const float* beta  = (const float*)d_in[4];
  float* out = (float*)d_out;

  char* ws = (char*)d_ws;
  size_t off = 0;
  const size_t xq_bytes = (size_t)B_ * HP * WP * CIN;
  signed char* xq = (signed char*)(ws + off); off += xq_bytes;
  off = (off + 255) & ~(size_t)255;
  signed char* bq = (signed char*)(ws + off); off += (size_t)COUT * KTOT;
  off = (off + 255) & ~(size_t)255;
  float* alpha = (float*)(ws + off); off += COUT * sizeof(float);
  off = (off + 255) & ~(size_t)255;
  float* gsum  = (float*)(ws + off); off += COUT * sizeof(float);
  off = (off + 255) & ~(size_t)255;
  float* gsq   = (float*)(ws + off); off += COUT * sizeof(float);
  off = (off + 255) & ~(size_t)255;
  float* yconv = (float*)(ws + off);

  (void)hipMemsetAsync(xq, 0, xq_bytes, stream);
  (void)hipMemsetAsync(gsum, 0, COUT * sizeof(float), stream);
  (void)hipMemsetAsync(gsq,  0, COUT * sizeof(float), stream);

  k_prep_w<<<COUT, 256, 0, stream>>>(w, bq, alpha);

  {
    size_t total = (size_t)B_ * CIN * HW;
    k_bin_x<<<(unsigned)((total + 255) / 256), 256, 0, stream>>>(x, xq);
  }

  k_conv<<<dim3(COUT / 16, M_TOT / 128), 256, 0, stream>>>(xq, bq, alpha, bias, yconv);

  k_stats<<<M_TOT / 256, 256, 0, stream>>>(yconv, gsum, gsq);

  {
    size_t total = (size_t)B_ * COUT * HW;
    k_final<<<(unsigned)((total + 255) / 256), 256, 0, stream>>>(yconv, gsum, gsq,
                                                                 gamma, beta, out);
  }
}
